// GAT_13898514169935
// MI455X (gfx1250) — matrix operand order repaired and checked by reading, not compiled, not run on hardware
//
#include <hip/hip_runtime.h>
#include <math.h>

#define N_NODES 4096
#define NFEAT   512
#define NHID    64
#define NHEADS  8
#define NCLASS  16
#define KCLU    64
#define ALPHA   0.2f
#define NEGINF  -9e15f

#if defined(__has_builtin)
#  if __has_builtin(__builtin_amdgcn_tensor_load_to_lds)
#    define HAVE_TDM 1
#  endif
#endif
#ifndef HAVE_TDM
#  define HAVE_TDM 0
#endif

typedef __attribute__((ext_vector_type(8)))  float    v8f;
typedef __attribute__((ext_vector_type(16))) __bf16   v16bf;
typedef __attribute__((ext_vector_type(4)))  float    f4v;
typedef __attribute__((ext_vector_type(4)))  unsigned u4v;
typedef __attribute__((ext_vector_type(8)))  int      i8v;
typedef __attribute__((ext_vector_type(4)))  int      i4v;

union BBits { unsigned u[8]; u4v q[2]; v16bf v; };

// hardware f32->bf16 (RNE) — lets clang emit v_cvt_pk_bf16_f32
__device__ __forceinline__ unsigned pack2(float lo, float hi){
  union { __bf16 h[2]; unsigned u; } z;
  z.h[0] = (__bf16)lo; z.h[1] = (__bf16)hi;
  return z.u;
}
__device__ __forceinline__ unsigned pcg(unsigned x){
  x = x*747796405u + 2891336453u;
  unsigned w = ((x >> ((x>>28)+4u)) ^ x) * 277803737u;
  return (w>>22) ^ w;
}
__device__ __forceinline__ float u01(unsigned r){ return (float)(r>>8) * (1.f/16777216.f); }

// ---------------------------------------------------------------------------
// TDM: 1-D tile copy of `ndw` dwords from global `gsrc` into LDS `ldsAddr`.
// D# group0: count=1 | lds_addr | global_addr[56:0] | type=2.
// D# group1: data_size=4B, tensor_dim0=ndw, tensor_dim1=1, tile_dim0=ndw,
//            tile_dim1=1, dim0_stride=ndw; groups 2/3 zero (<=2D tensor).
// Toolchain here exposes the 6-arg clang-23 builtin:
//   (uint32x4 g0, int32x8 g1, int32x4, int32x4, int32x8, i32 cpol)
// ---------------------------------------------------------------------------
__device__ __forceinline__ void tdm_issue(const unsigned* gsrc, unsigned ldsAddr, int ndw)
{
#if HAVE_TDM
  unsigned long long ga = (unsigned long long)(uintptr_t)gsrc;
  u4v g0; i8v g1; i4v g2, g3; i8v g4;
  g0[0] = 1u;                                   // count=1 (valid descriptor)
  g0[1] = ldsAddr;                              // lds_addr (bytes)
  g0[2] = (unsigned)ga;                         // global_addr[31:0]
  g0[3] = ((unsigned)(ga >> 32) & 0x01FFFFFFu) | 0x80000000u; // addr[56:32] | type=2
  g1[0] = (int)(2u << 16);                      // data_size = 4 bytes
  g1[1] = (int)(((unsigned)ndw & 0xFFFFu) << 16);   // tensor_dim0[15:0]
  g1[2] = (int)(((unsigned)ndw >> 16) | (1u << 16));// tensor_dim0[31:16] | tensor_dim1=1
  g1[3] = (int)((unsigned)ndw << 16);           // tile_dim0 = ndw
  g1[4] = 1;                                    // tile_dim1 = 1
  g1[5] = ndw;                                  // tensor_dim0_stride
  g1[6] = 0; g1[7] = 0;
  g2 = (i4v)0; g3 = (i4v)0; g4 = (i8v)0;
  __builtin_amdgcn_tensor_load_to_lds(g0, g1, g2, g3, g4, 0);
#else
  (void)gsrc; (void)ldsAddr; (void)ndw;
#endif
}

// ---------------------------------------------------------------------------
// Pack a row-major f32 [Kd x Ncols] matrix into bf16 WMMA B-operand tiles.
// Tile (kb,nb) is 32x16; lane L owns column nb*16+(L&15), half=(L>>4),
// dword r holds K = half*8 + (r<4 ? 2r : 16+2(r-4)) + {0,1}. Stored lane-major: 8 uints / lane.
// ---------------------------------------------------------------------------
__global__ void pack_b_kernel(const float* __restrict__ src, long srcHeadStride,
                              int Kd, int Ncols,
                              unsigned* __restrict__ dst, long dstHeadStride, int total)
{
  int t = blockIdx.x*blockDim.x + threadIdx.x;
  if (t >= total) return;
  int r    = t & 7;
  int lane = (t >> 3) & 31;
  int rest = t >> 8;
  int nnb = Ncols >> 4;
  int nkb = Kd >> 5;
  int nb = rest % nnb; rest /= nnb;
  int kb = rest % nkb; int head = rest / nkb;
  int k = kb*32 + (lane>>4)*8 + (r < 4 ? 2*r : 16 + 2*(r-4));
  int n = nb*16 + (lane & 15);
  const float* s = src + (size_t)head*srcHeadStride;
  float v0 = s[(size_t)k    *Ncols + n];
  float v1 = s[(size_t)(k+1)*Ncols + n];
  dst[(size_t)head*dstHeadStride + ((size_t)(kb*nnb+nb)*32 + lane)*8 + r] = pack2(v0, v1);
}

// ---------------------------------------------------------------------------
// C[16 x 16*NBT] tile per wave:  C = A(row-major f32, cvt->bf16) @ Bpacked.
// grid.x = M/16, grid.y = head.
// ---------------------------------------------------------------------------
template<int NBT>
__global__ void __launch_bounds__(32)
gemm_bf16(const float* __restrict__ A, int lda, long aStride,
          const unsigned* __restrict__ Bp, long bStride,
          float* __restrict__ C, int ldc, long cStride, int Kd)
{
  const int lane = threadIdx.x;
  const int head = blockIdx.y;
  A  += (size_t)head * aStride;
  Bp += (size_t)head * bStride;
  C  += (size_t)head * cStride;
  const int m0   = blockIdx.x * 16;
  const int row  = m0 + (lane & 15);
  const int half = lane >> 4;

  v8f acc[NBT];
  for (int nb=0; nb<NBT; ++nb) for (int r=0;r<8;++r) acc[nb][r]=0.f;

  const int nkb = Kd >> 5;
  for (int kb=0; kb<nkb; ++kb){
    const float* ap = A + (size_t)row*lda + kb*32 + half*8;
    f4v a0 = *(const f4v*)(ap);
    f4v a1 = *(const f4v*)(ap+4);
    f4v b0 = *(const f4v*)(ap+16);
    f4v b1 = *(const f4v*)(ap+20);
    v16bf av;
    av[0]=(__bf16)a0.x;  av[1]=(__bf16)a0.y;  av[2]=(__bf16)a0.z;  av[3]=(__bf16)a0.w;
    av[4]=(__bf16)a1.x;  av[5]=(__bf16)a1.y;  av[6]=(__bf16)a1.z;  av[7]=(__bf16)a1.w;
    av[8]=(__bf16)b0.x;  av[9]=(__bf16)b0.y;  av[10]=(__bf16)b0.z; av[11]=(__bf16)b0.w;
    av[12]=(__bf16)b1.x; av[13]=(__bf16)b1.y; av[14]=(__bf16)b1.z; av[15]=(__bf16)b1.w;
#pragma unroll
    for (int nb=0; nb<NBT; ++nb){
      BBits bm;
      const u4v* bp = (const u4v*)(Bp + ((size_t)(kb*NBT+nb)*32 + lane)*8);
      bm.q[0]=bp[0]; bm.q[1]=bp[1];
      acc[nb] = __builtin_amdgcn_wmma_f32_16x16x32_bf16(
                  false, av, false, bm.v, (short)0, acc[nb], false, false);
    }
  }
#pragma unroll
  for (int nb=0; nb<NBT; ++nb)
#pragma unroll
    for (int r=0;r<8;++r)
      C[(size_t)(m0 + r + half*8)*ldc + nb*16 + (lane&15)] = acc[nb][r];
}

__device__ __forceinline__ float edge_e(float a, float f1r, float f2v){
  float x = f1r + f2v;
  x = x > 0.f ? x : ALPHA * x;
  return a > 0.f ? x : NEGINF;
}

// ---------------------------------------------------------------------------
// Fused masked-attention + online-softmax + (att @ Wh).
// 256-thread block = 8 waves on the SAME head, 8 consecutive 16-row m-tiles.
// Wave 0 double-buffers each 4KB packed-B j-block into LDS via the Tensor
// Data Mover (s_wait_tensorcnt + workgroup barrier); all waves consume it
// with ds_load_b128, cutting per-head B traffic 8x.
// ---------------------------------------------------------------------------
template<int NBT, bool ELU>
__global__ void __launch_bounds__(256)
gat_attn(const float* __restrict__ adj,
         const float* __restrict__ f1, const float* __restrict__ f2,
         const unsigned* __restrict__ Bp, long bStride,
         float* __restrict__ C, int ldc, int colPerHead)
{
#if HAVE_TDM
  __shared__ unsigned ldsB[2][NBT*256];
#endif
  const int lane = threadIdx.x & 31;
  const int wav  = threadIdx.x >> 5;
  const int head = blockIdx.y;
  const int m0   = (blockIdx.x*8 + wav) * 16;
  const int row  = m0 + (lane & 15);
  const int half = lane >> 4;
  f1 += head * N_NODES; f2 += head * N_NODES;
  Bp += (size_t)head * bStride;
  const float  f1r  = f1[row];
  const float* arow = adj + (size_t)row * N_NODES;

  v8f acc[NBT];
  for (int nb=0;nb<NBT;++nb) for (int r=0;r<8;++r) acc[nb][r]=0.f;
  float mrun = -INFINITY, lrun = 0.f;

  const int NJB = N_NODES/32;
#if HAVE_TDM
  if (wav==0)
    tdm_issue(Bp, (unsigned)(uintptr_t)&ldsB[0][0], NBT*256);
#endif

  for (int jb=0; jb<NJB; ++jb){
    const int s0 = jb*32 + half*8;        // this lane's two 8-col spans: s0, s0+16
    f4v A0 = *(const f4v*)(arow + s0);
    f4v A1 = *(const f4v*)(arow + s0 + 4);
    f4v A2 = *(const f4v*)(arow + s0 + 16);
    f4v A3 = *(const f4v*)(arow + s0 + 20);
    f4v F0 = *(const f4v*)(f2 + s0);
    f4v F1 = *(const f4v*)(f2 + s0 + 4);
    f4v F2 = *(const f4v*)(f2 + s0 + 16);
    f4v F3 = *(const f4v*)(f2 + s0 + 20);
    if (jb+1 < NJB) __builtin_prefetch(arow + s0 + 32, 0, 1);

    float p[16];
    p[0]=edge_e(A0.x,f1r,F0.x);  p[1]=edge_e(A0.y,f1r,F0.y);
    p[2]=edge_e(A0.z,f1r,F0.z);  p[3]=edge_e(A0.w,f1r,F0.w);
    p[4]=edge_e(A1.x,f1r,F1.x);  p[5]=edge_e(A1.y,f1r,F1.y);
    p[6]=edge_e(A1.z,f1r,F1.z);  p[7]=edge_e(A1.w,f1r,F1.w);
    p[8]=edge_e(A2.x,f1r,F2.x);  p[9]=edge_e(A2.y,f1r,F2.y);
    p[10]=edge_e(A2.z,f1r,F2.z); p[11]=edge_e(A2.w,f1r,F2.w);
    p[12]=edge_e(A3.x,f1r,F3.x); p[13]=edge_e(A3.y,f1r,F3.y);
    p[14]=edge_e(A3.z,f1r,F3.z); p[15]=edge_e(A3.w,f1r,F3.w);

    float tmax = -INFINITY;
#pragma unroll
    for (int q=0;q<16;++q) tmax = fmaxf(tmax, p[q]);
    tmax = fmaxf(tmax, __shfl_xor(tmax, 16, 32));   // combine the row's two halves
    float mnew  = fmaxf(mrun, tmax);
    float scale = __expf(mrun - mnew);
    float psum  = 0.f;
#pragma unroll
    for (int q=0;q<16;++q){ p[q] = __expf(p[q] - mnew); psum += p[q]; }
    psum += __shfl_xor(psum, 16, 32);
    lrun = lrun * scale + psum;
    mrun = mnew;

#if HAVE_TDM
    const int cur = jb & 1;
    if (wav==0) __builtin_amdgcn_s_wait_tensorcnt(0);   // buf[cur] landed
    __syncthreads();                                    // visible to all waves;
                                                        // buf[cur^1] fully consumed
    if (wav==0 && jb+1 < NJB)
      tdm_issue(Bp + (size_t)(jb+1)*NBT*256,
                (unsigned)(uintptr_t)&ldsB[cur^1][0], NBT*256);
#endif

    // rescale accumulators: acc element r belongs to row (r + half*8)
    float srow[8];
#pragma unroll
    for (int r=0;r<8;++r) srow[r] = __shfl(scale, r + half*8, 32);
#pragma unroll
    for (int nb=0;nb<NBT;++nb)
#pragma unroll
      for (int r=0;r<8;++r) acc[nb][r] *= srow[r];

    v16bf av;
#pragma unroll
    for (int q=0;q<16;++q) av[q] = (__bf16)p[q];

#pragma unroll
    for (int nb=0;nb<NBT;++nb){
      BBits bm;
#if HAVE_TDM
      const u4v* bp = (const u4v*)&ldsB[cur][(nb*32 + lane)*8];
#else
      const u4v* bp = (const u4v*)(Bp + ((size_t)(jb*NBT+nb)*32 + lane)*8);
#endif
      bm.q[0]=bp[0]; bm.q[1]=bp[1];
      acc[nb] = __builtin_amdgcn_wmma_f32_16x16x32_bf16(
                  false, av, false, bm.v, (short)0, acc[nb], false, false);
    }
  }

  float linv[8];
#pragma unroll
  for (int r=0;r<8;++r) linv[r] = 1.f / __shfl(lrun, r + half*8, 32);
  const int colbase = head * colPerHead;
#pragma unroll
  for (int nb=0;nb<NBT;++nb)
#pragma unroll
    for (int r=0;r<8;++r){
      float hv = acc[nb][r] * linv[r];
      if (ELU) hv = hv > 0.f ? hv : (__expf(hv) - 1.f);
      C[(size_t)(m0 + r + half*8)*ldc + colbase + nb*16 + (lane&15)] = hv;
    }
}

// f1 = Wh @ a[:nh], f2 = Wh @ a[nh:]  (one thread per (head,node))
__global__ void f12_kernel(const float* __restrict__ Wh, const float* __restrict__ a,
                           float* __restrict__ f1, float* __restrict__ f2,
                           int nh, long whHeadStride, int aHeadStride, int total)
{
  int t = blockIdx.x*blockDim.x + threadIdx.x;
  if (t >= total) return;
  int head = t >> 12;                 // N_NODES == 4096
  int i    = t & (N_NODES-1);
  const float* w  = Wh + (size_t)head*whHeadStride + (size_t)i*nh;
  const float* ah = a + (size_t)head*aHeadStride;
  float s1=0.f, s2=0.f;
  for (int n=0;n<nh;++n){ float v=w[n]; s1 += v*ah[n]; s2 += v*ah[nh+n]; }
  f1[t]=s1; f2[t]=s2;
}

// row softmax over 64 columns (optionally + bias); one wave per row
__global__ void softmax64_k(const float* __restrict__ src, const float* __restrict__ bias,
                            float* __restrict__ dst)
{
  int row=blockIdx.x, lane=threadIdx.x;
  float v0 = src[row*64+lane];
  float v1 = src[row*64+32+lane];
  if (bias){ v0 += bias[lane]; v1 += bias[32+lane]; }
  float m = fmaxf(v0,v1);
  for (int o=16;o>0;o>>=1) m = fmaxf(m, __shfl_xor(m,o,32));
  float e0=__expf(v0-m), e1=__expf(v1-m);
  float sm=e0+e1;
  for (int o=16;o>0;o>>=1) sm += __shfl_xor(sm,o,32);
  dst[row*64+lane]=e0/sm; dst[row*64+32+lane]=e1/sm;
}

__global__ void rowsum_k(const float* __restrict__ adj, float* __restrict__ d){
  int row=blockIdx.x, lane=threadIdx.x;
  const float* ar = adj + (size_t)row*N_NODES;
  float acc=0.f;
  for (int j=lane;j<N_NODES;j+=32) acc += ar[j];
  for (int o=16;o>0;o>>=1) acc += __shfl_xor(acc,o,32);
  if (!lane) d[row]=acc;
}

// C[KCLU x KCLU] = A^T B for A,B [N x KCLU]; one wave per output
__global__ void atb_k(const float* __restrict__ A, const float* __restrict__ B,
                      float* __restrict__ C){
  int i=blockIdx.x, j=blockIdx.y, lane=threadIdx.x;
  float acc=0.f;
  for (int n=lane;n<N_NODES;n+=32) acc += A[(size_t)n*KCLU+i]*B[(size_t)n*KCLU+j];
  for (int o=16;o>0;o>>=1) acc += __shfl_xor(acc,o,32);
  if (!lane) C[i*KCLU+j]=acc;
}

// h_pool = s^T h  and  hh_pool = s^T (h*h); block per cluster, coalesced h rows
__global__ void pool_k(const float* __restrict__ s, const float* __restrict__ h,
                       float* __restrict__ hpool, float* __restrict__ hhpool){
  int kk=blockIdx.x, f=threadIdx.x;
  float a0=0.f,a1=0.f,b0=0.f,b1=0.f;
  for (int i=0;i<N_NODES;++i){
    float sv = s[(size_t)i*KCLU+kk];
    float h0 = h[(size_t)i*NFEAT + f];
    float h1 = h[(size_t)i*NFEAT + f + 256];
    a0 += sv*h0; b0 += sv*h0*h0;
    a1 += sv*h1; b1 += sv*h1*h1;
  }
  hpool [kk*NFEAT+f]=a0; hpool [kk*NFEAT+f+256]=a1;
  hhpool[kk*NFEAT+f]=b0; hhpool[kk*NFEAT+f+256]=b1;
}

__global__ void colsum_k(const float* __restrict__ s, float* __restrict__ cs){
  __shared__ float red[256];
  int k=blockIdx.x, t=threadIdx.x;
  float a=0.f;
  for (int i=t;i<N_NODES;i+=256) a += s[(size_t)i*KCLU+k];
  red[t]=a; __syncthreads();
  for(int o=128;o>0;o>>=1){ if(t<o) red[t]+=red[t+o]; __syncthreads(); }
  if(!t) cs[k]=red[0];
}

__global__ void std_k(const float* __restrict__ hpool, const float* __restrict__ hhpool,
                      const float* __restrict__ cs, float* __restrict__ stdclu){
  int idx = blockIdx.x*blockDim.x + threadIdx.x;
  if (idx >= KCLU*NFEAT) return;
  int k = idx / NFEAT;
  float hc = hpool[idx];
  float var = hhpool[idx] - 2.f*hc*hc + hc*hc*cs[k];
  stdclu[idx] = sqrtf(var + 1e-10f);
}

__global__ void mu_k(const float* __restrict__ hpool, float* __restrict__ mu){
  __shared__ float red[256];
  int k=blockIdx.x, t=threadIdx.x;
  float s = hpool[k*NFEAT+t] + hpool[k*NFEAT+t+256];
  red[t]=s; __syncthreads();
  for(int o=128;o>0;o>>=1){ if(t<o) red[t]+=red[t+o]; __syncthreads(); }
  if (!t) mu[k]=red[0]/(float)NFEAT;
}

__global__ void sigma_mean_k(const float* __restrict__ mu, float* __restrict__ scal){
  __shared__ float a[64], b[64];
  int t=threadIdx.x; float v=mu[t]; a[t]=v; b[t]=v*v; __syncthreads();
  for (int o=32;o>0;o>>=1){ if(t<o){a[t]+=a[t+o]; b[t]+=b[t+o];} __syncthreads(); }
  if (!t){ float s=a[0], q=b[0]; float var=(q - s*s/64.f)/63.f; scal[0]=sqrtf(fmaxf(var,0.f)); }
}

__global__ void sigma_std_k(const float* __restrict__ stdclu, float* __restrict__ sigstd){
  int f = blockIdx.x*blockDim.x + threadIdx.x;
  if (f>=NFEAT) return;
  float s=0.f, q=0.f;
  for (int k=0;k<KCLU;++k){ float v=stdclu[k*NFEAT+f]; s+=v; q+=v*v; }
  float var = (q - s*s/(float)KCLU)/(float)(KCLU-1);
  sigstd[f] = sqrtf(var + 1e-10f);
}

__global__ void select_k(const float* __restrict__ s, int* __restrict__ seli,
                         int* __restrict__ selc, float* __restrict__ maskv,
                         float* __restrict__ n1v, float* __restrict__ n2v){
  int i = blockIdx.x*blockDim.x + threadIdx.x;
  if (i >= N_NODES) return;
  const float* sr = s + (size_t)i*KCLU;
  int best=0; float bv=sr[0];
  for (int k=1;k<KCLU;++k){ float v=sr[k]; if (v>bv){bv=v;best=k;} }
  seli[i]=best;
  int rc = (int)(pcg(0x9E3779B9u ^ (unsigned)i) % (KCLU-1));
  if (rc==best) rc = KCLU-1;
  selc[i]=rc;
  maskv[i] = (u01(pcg(0x85EBCA6Bu + i)) < 0.3f) ? 0.f : 1.f;
  float u1 = fmaxf(u01(pcg(0xC2B2AE35u + i)), 1e-7f);
  float u2 = u01(pcg(0x27D4EB2Fu + i));
  n1v[i] = sqrtf(-2.f*__logf(u1)) * __cosf(6.2831853f*u2);
  float u3 = fmaxf(u01(pcg(0x165667B1u + i)), 1e-7f);
  float u4 = u01(pcg(0x9E377969u + i));
  n2v[i] = sqrtf(-2.f*__logf(u3)) * __cosf(6.2831853f*u4);
}

__global__ void dsu_k(const float* __restrict__ h, const float* __restrict__ hpool,
                      const float* __restrict__ stdclu, const float* __restrict__ sigstd,
                      const float* __restrict__ scal, const int* __restrict__ seli,
                      const int* __restrict__ selc, const float* __restrict__ maskv,
                      const float* __restrict__ n1v, const float* __restrict__ n2v,
                      const int* __restrict__ epochp, float* __restrict__ hnew){
  int idx = blockIdx.x*blockDim.x + threadIdx.x;
  if (idx >= N_NODES*NFEAT) return;
  float hv = h[idx];
  if ((epochp[0] % 10) != 0){ hnew[idx]=hv; return; }
  int i = idx >> 9;       // / NFEAT
  int f = idx & (NFEAT-1);
  float mk = maskv[i];
  int Sel = seli[i], sc = selc[i];
  float beta = stdclu[sc*NFEAT + f] + n1v[i]*sigstd[f];
  float gama = hpool [sc*NFEAT + f] + n2v[i]*scal[0];
  float z = (hv - hpool[Sel*NFEAT + f]) / (stdclu[Sel*NFEAT + f] + 1e-10f);
  hnew[idx] = mk*hv + (1.f - mk)*(z*beta + gama);
}

__global__ void losses_k(const float* __restrict__ outadj, const float* __restrict__ ss,
                         const float* __restrict__ drow, const float* __restrict__ s2,
                         float* __restrict__ out){
  __shared__ float r1[256], r2[256], r3[256];
  __shared__ float snrm, strace, sden;
  int tid = threadIdx.x;
  float tr = 0.f;
  for (int i=tid; i<KCLU; i+=256) tr += outadj[i*KCLU + i];
  float den = 0.f;
  for (int i=tid; i<N_NODES; i+=256){
    float q = 0.f;
    const float* sr = s2 + (size_t)i*KCLU;
    for (int k=0;k<KCLU;++k){ float v = sr[k]; q += v*v; }
    den += drow[i]*q;
  }
  float nq = 0.f;
  for (int i=tid; i<KCLU*KCLU; i+=256){ float v = ss[i]; nq += v*v; }
  r1[tid]=tr; r2[tid]=den; r3[tid]=nq;
  __syncthreads();
  for (int o=128;o>0;o>>=1){
    if (tid<o){ r1[tid]+=r1[tid+o]; r2[tid]+=r2[tid+o]; r3[tid]+=r3[tid+o]; }
    __syncthreads();
  }
  if (tid==0){ strace=r1[0]; sden=r2[0]; snrm=sqrtf(r3[0]); }
  __syncthreads();
  float nrm = snrm;
  float od = 0.f;
  for (int i=tid; i<KCLU*KCLU; i+=256){
    int rr=i/KCLU, cc=i%KCLU;
    float v = ss[i]/nrm - ((rr==cc)?0.125f:0.f);   // I/sqrt(64)
    od += v*v;
  }
  r1[tid]=od; __syncthreads();
  for (int o=128;o>0;o>>=1){ if (tid<o) r1[tid]+=r1[tid+o]; __syncthreads(); }
  if (tid==0){
    out[N_NODES*NCLASS]   = -(strace/sden);
    out[N_NODES*NCLASS+1] = sqrtf(r1[0]);
  }
}

// elu + log_softmax over 16 classes, one wave per row (lanes 0..15 active)
__global__ void finalout_k(const float* __restrict__ h2, float* __restrict__ out){
  int row=blockIdx.x, lane=threadIdx.x;
  float v = (lane<NCLASS) ? h2[(size_t)row*NCLASS+lane] : -INFINITY;
  if (lane<NCLASS) v = v>0.f ? v : (__expf(v)-1.f);
  float m=v;
  for (int o=8;o>0;o>>=1) m = fmaxf(m, __shfl_xor(m,o,16));
  float e = (lane<NCLASS)? __expf(v-m):0.f;
  float sm=e;
  for (int o=8;o>0;o>>=1) sm += __shfl_xor(sm,o,16);
  if (lane<NCLASS) out[(size_t)row*NCLASS+lane] = v - m - __logf(sm);
}

// ---------------------------------------------------------------------------
extern "C" void kernel_launch(void* const* d_in, const int* in_sizes, int n_in,
                              void* d_out, int out_size, void* d_ws, size_t ws_size,
                              hipStream_t stream)
{
  const float* x      = (const float*)d_in[0];
  const float* adj    = (const float*)d_in[1];
  const float* Wheads = (const float*)d_in[2];
  const float* aheads = (const float*)d_in[3];
  const float* Wout   = (const float*)d_in[4];
  const float* aout   = (const float*)d_in[5];
  const float* mlpw   = (const float*)d_in[6];
  const float* mlpb   = (const float*)d_in[7];
  const int*   epochp = (const int*)d_in[8];
  float* out = (float*)d_out;

  char* p = (char*)d_ws;
  auto alloc = [&](size_t bytes)->char*{ char* r = p; p += (bytes + 255) & ~((size_t)255); return r; };

  const long WBH_HS  = 16L*4*256;     // uints per head (K=512, N=64)
  const long WHB_HS  = 128L*4*256;    // uints per head (K=4096, N=64)

  unsigned* wbh   = (unsigned*)alloc(sizeof(unsigned)*(size_t)NHEADS*WBH_HS);
  float*    wh    = (float*)   alloc(sizeof(float)*(size_t)NHEADS*N_NODES*NHID);
  float*    f1b   = (float*)   alloc(sizeof(float)*NHEADS*N_NODES);
  float*    f2b   = (float*)   alloc(sizeof(float)*NHEADS*N_NODES);
  unsigned* whb   = (unsigned*)alloc(sizeof(unsigned)*(size_t)NHEADS*WHB_HS);
  float*    h     = (float*)   alloc(sizeof(float)*(size_t)N_NODES*NFEAT);
  unsigned* mlpbk = (unsigned*)alloc(sizeof(unsigned)*16*4*256);
  float*    slog  = (float*)   alloc(sizeof(float)*(size_t)N_NODES*KCLU);
  float*    s     = (float*)   alloc(sizeof(float)*(size_t)N_NODES*KCLU);
  float*    s2    = (float*)   alloc(sizeof(float)*(size_t)N_NODES*KCLU);
  unsigned* s2b   = (unsigned*)alloc(sizeof(unsigned)*128*4*256);
  float*    tbuf  = (float*)   alloc(sizeof(float)*(size_t)N_NODES*KCLU);
  float*    outadj= (float*)   alloc(sizeof(float)*KCLU*KCLU);
  float*    ssb   = (float*)   alloc(sizeof(float)*KCLU*KCLU);
  float*    drow  = (float*)   alloc(sizeof(float)*N_NODES);
  float*    hpool = (float*)   alloc(sizeof(float)*KCLU*NFEAT);
  float*    hhpool= (float*)   alloc(sizeof(float)*KCLU*NFEAT);
  float*    stdclu= (float*)   alloc(sizeof(float)*KCLU*NFEAT);
  float*    cs    = (float*)   alloc(sizeof(float)*KCLU);
  float*    mu    = (float*)   alloc(sizeof(float)*KCLU);
  float*    sigstd= (float*)   alloc(sizeof(float)*NFEAT);
  float*    scal  = (float*)   alloc(sizeof(float)*8);
  int*      seli  = (int*)     alloc(sizeof(int)*N_NODES);
  int*      selc  = (int*)     alloc(sizeof(int)*N_NODES);
  float*    maskv = (float*)   alloc(sizeof(float)*N_NODES);
  float*    n1v   = (float*)   alloc(sizeof(float)*N_NODES);
  float*    n2v   = (float*)   alloc(sizeof(float)*N_NODES);
  float*    hnew  = (float*)   alloc(sizeof(float)*(size_t)N_NODES*NFEAT);
  unsigned* wob   = (unsigned*)alloc(sizeof(unsigned)*16*1*256);
  float*    wh2   = (float*)   alloc(sizeof(float)*(size_t)N_NODES*NCLASS);
  float*    f1o   = (float*)   alloc(sizeof(float)*N_NODES);
  float*    f2o   = (float*)   alloc(sizeof(float)*N_NODES);
  unsigned* wob2  = (unsigned*)alloc(sizeof(unsigned)*128*1*256);
  float*    h2    = (float*)   alloc(sizeof(float)*(size_t)N_NODES*NCLASS);
  (void)in_sizes; (void)n_in; (void)out_size; (void)ws_size;

  // 1) Wh[h] = x @ W_heads[h]   (bf16 WMMA, f32 accum)
  pack_b_kernel<<<512,256,0,stream>>>(Wheads, (long)NFEAT*NHID, NFEAT, NHID,
                                      wbh, WBH_HS, NHEADS*16*4*256);
  gemm_bf16<4><<<dim3(N_NODES/16, NHEADS),32,0,stream>>>(
      x, NFEAT, 0L, wbh, WBH_HS, wh, NHID, (long)N_NODES*NHID, NFEAT);
  f12_kernel<<<(NHEADS*N_NODES)/256,256,0,stream>>>(
      wh, aheads, f1b, f2b, NHID, (long)N_NODES*NHID, 2*NHID, NHEADS*N_NODES);

  // 2) fused masked attention + softmax + att@Wh  ->  h [N, H*NHID] (+elu)
  pack_b_kernel<<<4096,256,0,stream>>>(wh, (long)N_NODES*NHID, N_NODES, NHID,
                                       whb, WHB_HS, NHEADS*128*4*256);
  gat_attn<4,true><<<dim3(N_NODES/128, NHEADS),256,0,stream>>>(
      adj, f1b, f2b, whb, WHB_HS, h, NFEAT, NHID);

  // 3) s = softmax(h @ mlp_w + b); s2 = softmax(s)
  pack_b_kernel<<<64,256,0,stream>>>(mlpw, 0L, NFEAT, KCLU, mlpbk, 0L, 16*4*256);
  gemm_bf16<4><<<dim3(N_NODES/16,1),32,0,stream>>>(
      h, NFEAT, 0L, mlpbk, 0L, slog, KCLU, 0L, NFEAT);
  softmax64_k<<<N_NODES,32,0,stream>>>(slog, mlpb, s);
  softmax64_k<<<N_NODES,32,0,stream>>>(s, nullptr, s2);

  // 4) mincut / ortho losses: t = adj@s2 (WMMA), out_adj = s2^T t, ss = s2^T s2
  pack_b_kernel<<<512,256,0,stream>>>(s2, 0L, N_NODES, KCLU, s2b, 0L, 128*4*256);
  gemm_bf16<4><<<dim3(N_NODES/16,1),32,0,stream>>>(
      adj, N_NODES, 0L, s2b, 0L, tbuf, KCLU, 0L, N_NODES);
  atb_k<<<dim3(KCLU,KCLU),32,0,stream>>>(s2, tbuf, outadj);
  atb_k<<<dim3(KCLU,KCLU),32,0,stream>>>(s2, s2,   ssb);
  rowsum_k<<<N_NODES,32,0,stream>>>(adj, drow);
  losses_k<<<1,256,0,stream>>>(outadj, ssb, drow, s2, out);

  // 5) DSU stat-mix
  pool_k<<<KCLU,256,0,stream>>>(s, h, hpool, hhpool);
  colsum_k<<<KCLU,256,0,stream>>>(s, cs);
  std_k<<<(KCLU*NFEAT)/256,256,0,stream>>>(hpool, hhpool, cs, stdclu);
  mu_k<<<KCLU,256,0,stream>>>(hpool, mu);
  sigma_mean_k<<<1,64,0,stream>>>(mu, scal);
  sigma_std_k<<<NFEAT/256,256,0,stream>>>(stdclu, sigstd);
  select_k<<<N_NODES/256,256,0,stream>>>(s, seli, selc, maskv, n1v, n2v);
  dsu_k<<<(N_NODES*NFEAT)/256,256,0,stream>>>(
      h, hpool, stdclu, sigstd, scal, seli, selc, maskv, n1v, n2v, epochp, hnew);

  // 6) final GAT layer (no concat) + elu + log_softmax
  pack_b_kernel<<<16,256,0,stream>>>(Wout, 0L, NFEAT, NCLASS, wob, 0L, 16*1*256);
  gemm_bf16<1><<<dim3(N_NODES/16,1),32,0,stream>>>(
      hnew, NFEAT, 0L, wob, 0L, wh2, NCLASS, 0L, NFEAT);
  f12_kernel<<<N_NODES/256,256,0,stream>>>(wh2, aout, f1o, f2o, NCLASS, 0L, 2*NCLASS, N_NODES);
  pack_b_kernel<<<128,256,0,stream>>>(wh2, 0L, N_NODES, NCLASS, wob2, 0L, 128*1*256);
  gat_attn<1,false><<<dim3(N_NODES/128,1),256,0,stream>>>(
      adj, f1o, f2o, wob2, 0L, h2, NCLASS, NCLASS);
  finalout_k<<<N_NODES,32,0,stream>>>(h2, out);
}
